// VideoModule_1898375545113
// MI455X (gfx1250) — matrix operand order repaired and checked by reading, not compiled, not run on hardware
//
#include <hip/hip_runtime.h>

#define DEV __device__ __forceinline__
#define USE_TDM 1   // GEMM tile staging: 1 = Tensor Data Mover, 0 = per-lane async

typedef __attribute__((ext_vector_type(16))) _Float16 v16h;
typedef __attribute__((ext_vector_type(8)))  _Float16 v8h;
typedef __attribute__((ext_vector_type(8)))  float    v8f;
typedef unsigned v4u __attribute__((ext_vector_type(4)));
typedef unsigned v8u __attribute__((ext_vector_type(8)));

constexpr int D_   = 1536;
constexpr int SV_  = 4096;
constexpr int SM_  = 1024;
constexpr int SC_  = 512;
constexpr int S_   = SV_ + SM_;   // 5120
constexpr int NH_  = 12;
constexpr int DH_  = 128;
constexpr int DFF_ = 6144;

// ---------------------------------------------------------------------------
// CDNA5 async global->LDS copy (ASYNCcnt), per CDNA5 ISA.
// ---------------------------------------------------------------------------
DEV void async_b128(unsigned ldsaddr, const void* gaddr) {
  asm volatile("global_load_async_to_lds_b128 %0, %1, off"
               :: "v"(ldsaddr), "v"(gaddr) : "memory");
}
DEV void wait_async_le4() { asm volatile("s_wait_asynccnt 0x4" ::: "memory"); }
DEV void wait_async_0()   { asm volatile("s_wait_asynccnt 0x0" ::: "memory"); }
DEV unsigned lds_addr_of(const void* p) {  // generic LDS ptr -> addr[31:0]
  return (unsigned)(uintptr_t)p;
}

// ---------------------------------------------------------------------------
// Tensor Data Mover (TENSORcnt), per CDNA5 ISA-8.
// 2-group descriptor (2D tensor): group0 = control/lds/global addr,
// group1 = dims/strides. One DMA moves a whole [tile1 x tile0] tile to LDS.
// ---------------------------------------------------------------------------
DEV void tdm_load(v4u g0, v8u g1) {
  asm volatile("tensor_load_to_lds %0, %1" :: "s"(g0), "s"(g1) : "memory");
}
DEV v4u tdm_g0(unsigned lds, unsigned long long ga) {
  v4u g;
  g.x = 1u;                                   // count=1, user mode
  g.y = lds;                                  // lds_addr (bytes)
  g.z = (unsigned)ga;                         // global_addr[31:0]
  g.w = ((unsigned)(ga >> 32) & 0x01ffffffu)  // global_addr[56:32]
        | (2u << 30);                         // type=2 ("image")
  return g;
}
// data_size=2B; 2D tensor dim0 x dim1 (stride0), tile tile0 x tile1
DEV v8u tdm_g1_2d(unsigned dim0, unsigned dim1, unsigned tile0, unsigned tile1,
                  unsigned stride0) {
  v8u g;
  g[0] = 0x00010000u;                                   // data_size=1 (2 bytes)
  g[1] = (dim0 & 0xffffu) << 16;                        // tensor_dim0[15:0]
  g[2] = ((dim0 >> 16) & 0xffffu) | ((dim1 & 0xffffu) << 16);
  g[3] = ((dim1 >> 16) & 0xffffu) | (tile0 << 16);      // tile_dim0
  g[4] = tile1;                                         // tile_dim1, tile_dim2=0
  g[5] = stride0;                                       // tensor_dim0_stride[31:0]
  g[6] = 0u;
  g[7] = 0u;
  return g;
}

// ---------------------------------------------------------------------------
// WMMA fragment helpers (CDNA5 ISA §7.12.2 layouts, wave32)
// ---------------------------------------------------------------------------
DEV v16h frag_load_a(const _Float16* base, int ld) {
  const int lane = threadIdx.x & 31;
  const _Float16* p = base + (lane & 15) * ld + ((lane >> 4) << 3);
  v8h lo = *(const v8h*)p;
  v8h hi = *(const v8h*)(p + 16);
  v16h r;
#pragma unroll
  for (int i = 0; i < 8; ++i) { r[i] = lo[i]; r[i + 8] = hi[i]; }
  return r;
}

DEV v16h frag_load_b(const _Float16* base, int ld) {
  const int lane = threadIdx.x & 31;
  const _Float16* p = base + (lane & 15) * ld + ((lane >> 4) << 3);
  v8h lo = *(const v8h*)p;
  v8h hi = *(const v8h*)(p + 16);
  v16h r;
#pragma unroll
  for (int i = 0; i < 8; ++i) { r[i] = lo[i]; r[i + 8] = hi[i]; }
  return r;
}

DEV v8f wmma32(v16h a, v16h b, v8f c) {
  return __builtin_amdgcn_wmma_f32_16x16x32_f16(false, a, false, b,
                                                (short)0, c, false, false);
}

DEV float gelu_tanh(float x) {
  float x3 = x * x * x;
  return 0.5f * x * (1.f + tanhf(0.7978845608028654f * (x + 0.044715f * x3)));
}

// ---------------------------------------------------------------------------
// Weight conversion: f32 [K][N] -> f16 transposed [N][K]
// ---------------------------------------------------------------------------
__global__ void k_transpose_f32_to_f16(const float* __restrict__ in,
                                       _Float16* __restrict__ out,
                                       int K, int N) {
  size_t i = (size_t)blockIdx.x * blockDim.x + threadIdx.x;
  size_t total = (size_t)K * N;
  if (i >= total) return;
  int k = (int)(i / N);
  int n = (int)(i - (size_t)k * N);
  out[(size_t)n * K + k] = (_Float16)in[i];
}

__global__ void k_f32_to_f16(const float* __restrict__ in,
                             _Float16* __restrict__ out, size_t n) {
  size_t i = (size_t)blockIdx.x * blockDim.x + threadIdx.x;
  if (i < n) out[i] = (_Float16)in[i];
}

// ---------------------------------------------------------------------------
// Fused LayerNorm (+ optional AdaLN modulate) -> f16
// ---------------------------------------------------------------------------
__global__ __launch_bounds__(256) void k_mod_ln_f16(
    const float* __restrict__ x,
    const float* __restrict__ tmod,    // null -> plain LN
    const float* __restrict__ mp,
    int is_, int ic_,
    _Float16* __restrict__ out)
{
  const int t = blockIdx.x;
  const float* xr = x + (size_t)t * D_;
  float lv[6];
  float s0 = 0.f, s1 = 0.f;
#pragma unroll
  for (int j = 0; j < 6; ++j) {
    float v = xr[threadIdx.x + j * 256];
    lv[j] = v; s0 += v; s1 += v * v;
  }
#pragma unroll
  for (int o = 16; o > 0; o >>= 1) {
    s0 += __shfl_xor(s0, o, 32);
    s1 += __shfl_xor(s1, o, 32);
  }
  __shared__ float rs0[8], rs1[8], stat[2];
  const int w = threadIdx.x >> 5, lane = threadIdx.x & 31;
  if (lane == 0) { rs0[w] = s0; rs1[w] = s1; }
  __syncthreads();
  if (threadIdx.x == 0) {
    float a = 0.f, b = 0.f;
#pragma unroll
    for (int j = 0; j < 8; ++j) { a += rs0[j]; b += rs1[j]; }
    float mean = a * (1.f / D_);
    float var  = b * (1.f / D_) - mean * mean;
    stat[0] = mean;
    stat[1] = rsqrtf(var + 1e-6f);
  }
  __syncthreads();
  const float mean = stat[0], rstd = stat[1];
#pragma unroll
  for (int j = 0; j < 6; ++j) {
    int d = threadIdx.x + j * 256;
    float v = (lv[j] - mean) * rstd;
    if (tmod) {
      float sh = mp[(size_t)is_ * D_ + d] + tmod[((size_t)t * 6 + is_) * D_ + d];
      float sc = mp[(size_t)ic_ * D_ + d] + tmod[((size_t)t * 6 + ic_) * D_ + d];
      v = v * (1.f + sc) + sh;
    }
    out[(size_t)t * D_ + d] = (_Float16)v;
  }
}

// ---------------------------------------------------------------------------
// RoPE in-place on f16 Q and K: one block per token
// ---------------------------------------------------------------------------
__global__ __launch_bounds__(256) void k_rope_f16(
    _Float16* __restrict__ q, _Float16* __restrict__ k,
    const float* __restrict__ freqs)
{
  const int t = blockIdx.x;
  for (int p = threadIdx.x; p < NH_ * (DH_ / 2); p += 256) {
    int h = p >> 6, i = p & 63;
    float f = freqs[(size_t)t * 64 + i];
    float c = cosf(f), s = sinf(f);
    size_t off = (size_t)t * D_ + h * DH_ + 2 * i;
    float q1 = (float)q[off], q2 = (float)q[off + 1];
    q[off]     = (_Float16)(q1 * c - q2 * s);
    q[off + 1] = (_Float16)(q1 * s + q2 * c);
    float k1 = (float)k[off], k2 = (float)k[off + 1];
    k[off]     = (_Float16)(k1 * c - k2 * s);
    k[off + 1] = (_Float16)(k1 * s + k2 * c);
  }
}

// ---------------------------------------------------------------------------
// WMMA GEMM: C[M][N] = act(A[M][K] * Bt[N][K]^T + bias)
// 256 threads = 8 waves; 128x128 tile; K-step 32; double-buffered LDS.
// Tile staging via TDM (one descriptor per tile, issued by wave 0) or
// per-lane async copies. Last K-step peeled for a branch-free steady state.
// flags: bit0 = f16 output, bit1 = tanh-GELU
// ---------------------------------------------------------------------------
__global__ __launch_bounds__(256) void k_gemm_f16_nt(
    const _Float16* __restrict__ A, const _Float16* __restrict__ Bt,
    void* __restrict__ Cout, const float* __restrict__ bias,
    int M, int N, int K, int flags)
{
  __shared__ __align__(16) _Float16 As[2][128 * 32];
  __shared__ __align__(16) _Float16 Bs[2][128 * 32];
  const int tm = blockIdx.y * 128;
  const int tn = blockIdx.x * 128;
  const int tid = threadIdx.x;
  const int wave = tid >> 5;
  const int lane = tid & 31;
  const int wm = (wave & 3) * 32;
  const int wn = (wave >> 2) * 64;

  v8f zero = {};
  v8f acc[2][4];
#pragma unroll
  for (int i = 0; i < 2; ++i)
#pragma unroll
    for (int j = 0; j < 4; ++j) acc[i][j] = zero;

  const int nk = K >> 5;

#if USE_TDM
  const unsigned long long gaA =
      (unsigned long long)(uintptr_t)A + ((size_t)tm * K) * 2;
  const unsigned long long gaB =
      (unsigned long long)(uintptr_t)Bt + ((size_t)tn * K) * 2;
  const v8u g1 = tdm_g1_2d((unsigned)K, 128u, 32u, 128u, (unsigned)K);
  const unsigned ldsA[2] = { lds_addr_of(&As[0][0]), lds_addr_of(&As[1][0]) };
  const unsigned ldsB[2] = { lds_addr_of(&Bs[0][0]), lds_addr_of(&Bs[1][0]) };
  if (wave == 0) {                       // stage tile 0 (two DMA descriptors)
    tdm_load(tdm_g0(ldsA[0], gaA), g1);
    tdm_load(tdm_g0(ldsB[0], gaB), g1);
  }
#else
  const int lr = tid >> 1;
  const int lc = (tid & 1) << 4;
  const _Float16* ga = A  + (size_t)(tm + lr) * K + lc;
  const _Float16* gb = Bt + (size_t)(tn + lr) * K + lc;
  const unsigned laA[2] = { lds_addr_of(&As[0][lr * 32 + lc]),
                            lds_addr_of(&As[1][lr * 32 + lc]) };
  const unsigned laB[2] = { lds_addr_of(&Bs[0][lr * 32 + lc]),
                            lds_addr_of(&Bs[1][lr * 32 + lc]) };
  async_b128(laA[0], ga);      async_b128(laA[0] + 16, ga + 8);
  async_b128(laB[0], gb);      async_b128(laB[0] + 16, gb + 8);
#endif

  auto compute = [&](int buf) {
    v16h a0 = frag_load_a(&As[buf][wm * 32], 32);
    v16h a1 = frag_load_a(&As[buf][(wm + 16) * 32], 32);
#pragma unroll
    for (int j = 0; j < 4; ++j) {
      v16h b = frag_load_b(&Bs[buf][(wn + j * 16) * 32], 32);
      acc[0][j] = wmma32(a0, b, acc[0][j]);
      acc[1][j] = wmma32(a1, b, acc[1][j]);
    }
  };

  for (int it = 0; it < nk - 1; ++it) {    // steady state: stage it+1, run it
    const int buf = it & 1;
#if USE_TDM
    if (wave == 0) {
      const unsigned long long koff = (unsigned long long)((it + 1) << 5) * 2;
      tdm_load(tdm_g0(ldsA[buf ^ 1], gaA + koff), g1);
      tdm_load(tdm_g0(ldsB[buf ^ 1], gaB + koff), g1);
      __builtin_amdgcn_s_wait_tensorcnt(2);   // oldest 2 (tile it) complete
    }
#else
    const int k1 = (it + 1) << 5;
    async_b128(laA[buf ^ 1], ga + k1);  async_b128(laA[buf ^ 1] + 16, ga + k1 + 8);
    async_b128(laB[buf ^ 1], gb + k1);  async_b128(laB[buf ^ 1] + 16, gb + k1 + 8);
    wait_async_le4();
#endif
    __syncthreads();                     // tile it visible to all waves
    compute(buf);
    __syncthreads();                     // reads done before buf reuse
  }
  {                                      // peeled final K-step
    const int buf = (nk - 1) & 1;
#if USE_TDM
    if (wave == 0) __builtin_amdgcn_s_wait_tensorcnt(0);
#else
    wait_async_0();
#endif
    __syncthreads();
    compute(buf);
  }

  const int cn = lane & 15;
  const int rm = (lane >> 4) << 3;
#pragma unroll
  for (int i = 0; i < 2; ++i) {
#pragma unroll
    for (int j = 0; j < 4; ++j) {
      const int n = tn + wn + j * 16 + cn;
      const float bv = bias ? bias[n] : 0.f;
      const int mbase = tm + wm + i * 16 + rm;
#pragma unroll
      for (int r = 0; r < 8; ++r) {
        float x = acc[i][j][r] + bv;
        if (flags & 2) x = gelu_tanh(x);
        size_t off = (size_t)(mbase + r) * N + n;
        if (flags & 1) ((_Float16*)Cout)[off] = (_Float16)x;
        else           ((float*)Cout)[off] = x;
      }
    }
  }
}

// ---------------------------------------------------------------------------
// Flash attention (one block = one 64-query tile of one head, 8 waves)
// ---------------------------------------------------------------------------
__global__ __launch_bounds__(256) void k_flash_attn(
    const _Float16* __restrict__ Q, const _Float16* __restrict__ K,
    const _Float16* __restrict__ V, const unsigned char* __restrict__ mask,
    _Float16* __restrict__ O, int Skv, float scale)
{
  const int h  = blockIdx.y;
  const int q0 = blockIdx.x * 64;
  const int tid  = threadIdx.x;
  const int wave = tid >> 5;
  const int lane = tid & 31;
  const int mq   = wave & 3;
  const int sgrp = wave >> 2;

  __shared__ __align__(16) _Float16 Ks[64 * 128];
  __shared__ __align__(16) _Float16 Vts[128 * 64];
  __shared__ __align__(16) float    Ss[64 * 64];
  __shared__ __align__(16) _Float16 Ps[64 * 64];
  __shared__ float row_m[64], row_l[64], row_scale[64];

  v16h qf[4];
  {
    const _Float16* qb = Q + (size_t)(q0 + mq * 16) * D_ + h * DH_;
#pragma unroll
    for (int ks = 0; ks < 4; ++ks) qf[ks] = frag_load_a(qb + ks * 32, D_);
  }

  v8f zero = {};
  v8f oacc[4] = {zero, zero, zero, zero};

  if (tid < 64) { row_m[tid] = -3.4e38f; row_l[tid] = 0.f; }
  __syncthreads();

  const int lr = tid >> 2;
  const int lc = (tid & 3) * 32;
  const unsigned lk = lds_addr_of(&Ks[lr * 128 + lc]);

  for (int kv0 = 0; kv0 < Skv; kv0 += 64) {
    { // K chunk via async copy; V chunk transposed via VGPRs
      const _Float16* kg = K + (size_t)(kv0 + lr) * D_ + h * DH_ + lc;
#pragma unroll
      for (int u = 0; u < 4; ++u)
        async_b128(lk + u * 16, kg + u * 8);
      const _Float16* vg = V + (size_t)(kv0 + lr) * D_ + h * DH_ + lc;
#pragma unroll
      for (int u = 0; u < 32; ++u)
        Vts[(lc + u) * 64 + lr] = vg[u];
    }
    wait_async_0();
    __syncthreads();

    v8f sacc0 = zero, sacc1 = zero;
#pragma unroll
    for (int ks = 0; ks < 4; ++ks) {
      v16h b0 = frag_load_b(&Ks[(sgrp * 32 + 0)  * 128 + ks * 32], 128);
      v16h b1 = frag_load_b(&Ks[(sgrp * 32 + 16) * 128 + ks * 32], 128);
      sacc0 = wmma32(qf[ks], b0, sacc0);
      sacc1 = wmma32(qf[ks], b1, sacc1);
    }
    {
      const int n0 = sgrp * 32 + (lane & 15);
      const int m0 = mq * 16 + ((lane >> 4) << 3);
#pragma unroll
      for (int r = 0; r < 8; ++r) {
        Ss[(m0 + r) * 64 + n0]      = sacc0[r];
        Ss[(m0 + r) * 64 + n0 + 16] = sacc1[r];
      }
    }
    __syncthreads();

    {
      const int row = tid >> 2, seg = tid & 3;
      float vals[16];
      float mloc = -3.4e38f;
#pragma unroll
      for (int u = 0; u < 16; ++u) {
        int n = seg * 16 + u;
        float s = Ss[row * 64 + n] * scale;
        if (mask && !mask[kv0 + n]) s = -1e9f;
        vals[u] = s;
        mloc = fmaxf(mloc, s);
      }
      mloc = fmaxf(mloc, __shfl_xor(mloc, 1, 32));
      mloc = fmaxf(mloc, __shfl_xor(mloc, 2, 32));
      const float newm = fmaxf(row_m[row], mloc);
      float lloc = 0.f;
#pragma unroll
      for (int u = 0; u < 16; ++u) {
        float e = __expf(vals[u] - newm);
        lloc += e;
        Ps[row * 64 + seg * 16 + u] = (_Float16)e;
      }
      lloc += __shfl_xor(lloc, 1, 32);
      lloc += __shfl_xor(lloc, 2, 32);
      if (seg == 0) {
        float rs = __expf(row_m[row] - newm);
        row_scale[row] = rs;
        row_l[row] = row_l[row] * rs + lloc;
        row_m[row] = newm;
      }
    }
    __syncthreads();

    {
      const int m0 = mq * 16 + ((lane >> 4) << 3);
#pragma unroll
      for (int r = 0; r < 8; ++r) {
        float rs = row_scale[m0 + r];
#pragma unroll
        for (int j = 0; j < 4; ++j) oacc[j][r] *= rs;
      }
#pragma unroll
      for (int ks = 0; ks < 2; ++ks) {
        v16h pa = frag_load_a(&Ps[(mq * 16) * 64 + ks * 32], 64);
#pragma unroll
        for (int j = 0; j < 4; ++j) {
          int nt = sgrp * 4 + j;
          v16h vb = frag_load_b(&Vts[(nt * 16) * 64 + ks * 32], 64);
          oacc[j] = wmma32(pa, vb, oacc[j]);
        }
      }
    }
    __syncthreads();
  }

  {
    const int m0 = mq * 16 + ((lane >> 4) << 3);
    const int nlane = lane & 15;
#pragma unroll
    for (int r = 0; r < 8; ++r) {
      float inv = 1.f / fmaxf(row_l[m0 + r], 1e-20f);
#pragma unroll
      for (int j = 0; j < 4; ++j) {
        int n = (sgrp * 4 + j) * 16 + nlane;
        O[(size_t)(q0 + m0 + r) * D_ + h * DH_ + n] = (_Float16)(oacc[j][r] * inv);
      }
    }
  }
}

// ---------------------------------------------------------------------------
// out = base + delta * (mp[idx] + tmod[:,idx])   (mp==null -> plain add)
// ---------------------------------------------------------------------------
__global__ void k_residual_modscale(
    const float* __restrict__ base, const float* __restrict__ delta,
    const float* __restrict__ mp, const float* __restrict__ tmod, int idx,
    float* __restrict__ out, int rows)
{
  size_t i = (size_t)blockIdx.x * blockDim.x + threadIdx.x;
  size_t total = (size_t)rows * D_;
  if (i >= total) return;
  int t = (int)(i / D_);
  int d = (int)(i - (size_t)t * D_);
  float g = 1.f;
  if (mp) g = mp[(size_t)idx * D_ + d] + tmod[((size_t)t * 6 + idx) * D_ + d];
  out[i] = base[i] + delta[i] * g;
}

// ---------------------------------------------------------------------------
// Host orchestration
// ---------------------------------------------------------------------------
extern "C" void kernel_launch(void* const* d_in, const int* in_sizes, int n_in,
                              void* d_out, int out_size, void* d_ws, size_t ws_size,
                              hipStream_t stream) {
  (void)in_sizes; (void)n_in; (void)out_size; (void)ws_size;

  const float* video_latent = (const float*)d_in[0];
  const float* modal_latent = (const float*)d_in[1];
  const float* context      = (const float*)d_in[2];
  const unsigned char* ctx_mask = (const unsigned char*)d_in[3];
  const float* video_t_mod  = (const float*)d_in[4];
  const float* modal_t_mod  = (const float*)d_in[5];
  const float* video_freqs  = (const float*)d_in[6];
  const float* modal_freqs  = (const float*)d_in[7];
  const float* mod_param_v  = (const float*)d_in[8];
  const float* mod_param_m  = (const float*)d_in[9];
  const float* b1 = (const float*)d_in[23];
  const float* b2 = (const float*)d_in[25];
  float* out_video = (float*)d_out;
  float* out_modal = (float*)d_out + (size_t)SV_ * D_;

  char* wp = (char*)d_ws;
  auto alloc = [&](size_t bytes) -> void* {
    void* p = wp;
    wp += (bytes + 255) & ~(size_t)255;
    return p;
  };
  _Float16* wt[12];
  for (int i = 0; i < 12; ++i) wt[i] = (_Float16*)alloc((size_t)D_ * D_ * 2);
  _Float16* w1t   = (_Float16*)alloc((size_t)DFF_ * D_ * 2);
  _Float16* w2t   = (_Float16*)alloc((size_t)D_ * DFF_ * 2);
  _Float16* Xh    = (_Float16*)alloc((size_t)S_ * D_ * 2);
  _Float16* Qh    = (_Float16*)alloc((size_t)S_ * D_ * 2);
  _Float16* Kh    = (_Float16*)alloc((size_t)S_ * D_ * 2);
  _Float16* Vh    = (_Float16*)alloc((size_t)S_ * D_ * 2);
  _Float16* Oh    = (_Float16*)alloc((size_t)S_ * D_ * 2);
  float*    P32   = (float*)alloc((size_t)S_ * D_ * 4);
  float*    videoF= (float*)alloc((size_t)SV_ * D_ * 4);
  _Float16* ctx16 = (_Float16*)alloc((size_t)SC_ * D_ * 2);
  _Float16* ffnh  = (_Float16*)alloc((size_t)SV_ * DFF_ * 2);

  auto gemm = [&](const _Float16* A, const _Float16* Bt, void* C,
                  const float* bias, int M, int N, int K, int flags) {
    dim3 g(N / 128, M / 128);
    k_gemm_f16_nt<<<g, 256, 0, stream>>>(A, Bt, C, bias, M, N, K, flags);
  };
  const float att_scale = 0.08838834764831845f;  // 1/sqrt(128)

  // --- 0) weight conversion
  {
    int blks = (D_ * D_ + 255) / 256;
    for (int i = 0; i < 12; ++i)
      k_transpose_f32_to_f16<<<blks, 256, 0, stream>>>((const float*)d_in[10 + i], wt[i], D_, D_);
    int blks1 = (D_ * DFF_ + 255) / 256;
    k_transpose_f32_to_f16<<<blks1, 256, 0, stream>>>((const float*)d_in[22], w1t, D_, DFF_);
    k_transpose_f32_to_f16<<<blks1, 256, 0, stream>>>((const float*)d_in[24], w2t, DFF_, D_);
    k_f32_to_f16<<<(SC_ * D_ + 255) / 256, 256, 0, stream>>>(context, ctx16, (size_t)SC_ * D_);
  }

  // --- 1) AdaLN LN+modulate (shift=chunk0, scale=chunk1)
  k_mod_ln_f16<<<SV_, 256, 0, stream>>>(video_latent, video_t_mod, mod_param_v, 0, 1, Xh);
  k_mod_ln_f16<<<SM_, 256, 0, stream>>>(modal_latent, modal_t_mod, mod_param_m, 0, 1, Xh + (size_t)SV_ * D_);

  // --- 2) QKV projections
  gemm(Xh,                    wt[0], Qh,                    nullptr, SV_, D_, D_, 1);
  gemm(Xh + (size_t)SV_ * D_, wt[4], Qh + (size_t)SV_ * D_, nullptr, SM_, D_, D_, 1);
  gemm(Xh,                    wt[1], Kh,                    nullptr, SV_, D_, D_, 1);
  gemm(Xh + (size_t)SV_ * D_, wt[5], Kh + (size_t)SV_ * D_, nullptr, SM_, D_, D_, 1);
  gemm(Xh,                    wt[2], Vh,                    nullptr, SV_, D_, D_, 1);
  gemm(Xh + (size_t)SV_ * D_, wt[6], Vh + (size_t)SV_ * D_, nullptr, SM_, D_, D_, 1);

  // --- 3) RoPE
  k_rope_f16<<<SV_, 256, 0, stream>>>(Qh, Kh, video_freqs);
  k_rope_f16<<<SM_, 256, 0, stream>>>(Qh + (size_t)SV_ * D_, Kh + (size_t)SV_ * D_, modal_freqs);

  // --- 4) joint self-attention
  {
    dim3 g(S_ / 64, NH_);
    k_flash_attn<<<g, 256, 0, stream>>>(Qh, Kh, Vh, nullptr, Oh, S_, att_scale);
  }

  // --- 5) output projections
  gemm(Oh,                    wt[3], P32,                    nullptr, SV_, D_, D_, 0);
  gemm(Oh + (size_t)SV_ * D_, wt[7], P32 + (size_t)SV_ * D_, nullptr, SM_, D_, D_, 0);

  // --- 6) residuals
  {
    int blksm = (SM_ * D_ + 255) / 256;
    k_residual_modscale<<<blksm, 256, 0, stream>>>(modal_latent, P32 + (size_t)SV_ * D_,
                                                   mod_param_m, modal_t_mod, 2, out_modal, SM_);
    int blksv = (SV_ * D_ + 255) / 256;
    k_residual_modscale<<<blksv, 256, 0, stream>>>(video_latent, P32,
                                                   mod_param_v, video_t_mod, 2, videoF, SV_);
  }

  // --- 7) cross attention
  k_mod_ln_f16<<<SV_, 256, 0, stream>>>(videoF, nullptr, nullptr, 0, 0, Xh);
  gemm(Xh,    wt[8],  Qh, nullptr, SV_, D_, D_, 1);
  gemm(ctx16, wt[9],  Kh, nullptr, SC_, D_, D_, 1);
  gemm(ctx16, wt[10], Vh, nullptr, SC_, D_, D_, 1);
  {
    dim3 g(SV_ / 64, NH_);
    k_flash_attn<<<g, 256, 0, stream>>>(Qh, Kh, Vh, ctx_mask, Oh, SC_, att_scale);
  }
  gemm(Oh, wt[11], P32, nullptr, SV_, D_, D_, 0);
  {
    int blksv = (SV_ * D_ + 255) / 256;
    k_residual_modscale<<<blksv, 256, 0, stream>>>(videoF, P32, nullptr, nullptr, 0, videoF, SV_);
  }

  // --- 8) FFN
  k_mod_ln_f16<<<SV_, 256, 0, stream>>>(videoF, video_t_mod, mod_param_v, 3, 4, Xh);
  gemm(Xh,   w1t, ffnh, b1, SV_, DFF_, D_, 1 | 2);
  gemm(ffnh, w2t, P32,  b2, SV_, D_, DFF_, 0);
  {
    int blksv = (SV_ * D_ + 255) / 256;
    k_residual_modscale<<<blksv, 256, 0, stream>>>(videoF, P32, mod_param_v, video_t_mod, 5,
                                                   out_video, SV_);
  }
}
